// LocalPatternAware_21174188769888
// MI455X (gfx1250) — matrix operand order repaired and checked by reading, not compiled, not run on hardware
//
#include <hip/hip_runtime.h>

typedef __attribute__((ext_vector_type(16))) __bf16 v16bf;
typedef __attribute__((ext_vector_type(8)))  __bf16 v8bf;
typedef __attribute__((ext_vector_type(8)))  float  v8f;
typedef __attribute__((ext_vector_type(4)))  unsigned int v4u;
typedef __attribute__((ext_vector_type(4)))  int v4i;
typedef __attribute__((ext_vector_type(8)))  int v8i;

#define C_DIM 256
#define HW    9216
#define LDT   40   // bf16 X-tile pitch in elements (80 B, multiple of 16 B)
#define LDW   36   // f32 W-tile pitch in elements (144 B): 32 cols + 4 pad dwords

#if defined(__HIP_DEVICE_COMPILE__) &&                                     \
    __has_builtin(__builtin_amdgcn_tensor_load_to_lds) &&                  \
    __has_builtin(__builtin_amdgcn_s_wait_tensorcnt)
#define HAVE_TDM 1
#else
#define HAVE_TDM 0
#endif

#if HAVE_TDM
// ---------------------------------------------------------------------------
// TDM: DMA a 2D f32 tile [tile_dim1=128 rows x tile_dim0=32 cols] with row
// stride 256 elements from global into LDS, padding +4 dwords after every
// 32 dwords (=> 36-dword LDS row pitch, avoids 16-way bank conflicts).
// Descriptor bitfields per CDNA5 ISA.
// ---------------------------------------------------------------------------
__device__ __forceinline__ void tdm_load_w_tile(const float* gsrc, void* lds_dst)
{
    const unsigned long long ga = (unsigned long long)(size_t)gsrc;
    const unsigned lds = (unsigned)(size_t)lds_dst;   // generic addr low 32 = LDS offset
    v4u g0;
    g0.x = 1u;                                           // count=1 (valid user D#)
    g0.y = lds;                                          // lds_addr (bytes)
    g0.z = (unsigned)(ga & 0xffffffffu);                 // global_addr[31:0]
    g0.w = (unsigned)((ga >> 32) & 0x01ffffffu)          // global_addr[56:32]
         | (2u << 30);                                   // type=2 ("image")
    v8i g1;
    // workgroup_mask=0 | data_size=2 (4B) | pad_enable | pad_interval=4 (32 dw)
    // | pad_amount=3 (4 dw)
    g1[0] = 0x20000 | (1 << 20) | (4 << 22) | (3 << 25);
    g1[1] = (int)((C_DIM & 0xffff) << 16);               // tensor_dim0[15:0]
    g1[2] = (int)((C_DIM >> 16) | ((C_DIM & 0xffff) << 16)); // dim0 hi | dim1 lo
    g1[3] = (int)((C_DIM >> 16) | (32u << 16));          // dim1 hi | tile_dim0=32
    g1[4] = 128;                                         // tile_dim1=128, tile_dim2=0
    g1[5] = C_DIM;                                       // tensor_dim0_stride[31:0]
    g1[6] = 0;                                           // stride hi | dim1_stride lo
    g1[7] = 0;
    const v4i z4 = {0, 0, 0, 0};
#if __has_include(<hip/amd_detail/amd_gfx1250_TDM.h>)
    const v8i z8 = {0, 0, 0, 0, 0, 0, 0, 0};
    __builtin_amdgcn_tensor_load_to_lds(g0, g1, z4, z4, z8, 0);   // clang-23 form
#else
    __builtin_amdgcn_tensor_load_to_lds(g0, g1, z4, z4, 0);       // ROCm 7.2 form
#endif
}
#endif

// ---------------------------------------------------------------------------
// WMMA GEMM: Y[d, p] = sum_c W[d, c] * X[img, c, p]   (+bias, *scale)
// EPI==0: store bf16 to Ybf[img*C*HW + d*HW + p]   (projection outputs)
// EPI==1: store f32  to Yf[(p*2 + img)*C + d]      (final [hw, bs, c] output)
// Block: 256 threads (8 waves). Tile: 128 d x 128 p. K loop: 8 x 32.
// W tile staged via TDM (f32, padded pitch); X tile staged transposed as bf16.
// ---------------------------------------------------------------------------
template <int EPI>
__global__ __launch_bounds__(256) void gemm_wmma(
    const float* __restrict__ X, const float* __restrict__ Wg,
    const float* __restrict__ bias, __bf16* __restrict__ Ybf,
    float* __restrict__ Yf, float scale)
{
    __shared__ float  Wf[128 * LDW];  // Wf[d][c] f32 (TDM-padded pitch)
    __shared__ __bf16 Xt[128 * LDT];  // Xt[p][c] bf16 (transposed stage)

    const int tid   = threadIdx.x;
    const int pBase = blockIdx.x * 128;
    const int dBase = blockIdx.y * 128;
    const int img   = blockIdx.z;
    const int wv    = tid >> 5;
    const int lane  = tid & 31;
    const int half  = (lane >> 4) & 1;
    const int l16   = lane & 15;

    const float* Ximg = X + (size_t)img * C_DIM * HW;

    v8f acc[8];
    const v8f vzero = {0.f, 0.f, 0.f, 0.f, 0.f, 0.f, 0.f, 0.f};
#pragma unroll
    for (int s = 0; s < 8; ++s) acc[s] = vzero;

    for (int k0 = 0; k0 < C_DIM; k0 += 32) {
        // ---- stage W tile: 128 rows x 32 cols f32 --------------------------
#if HAVE_TDM
        if (wv == 0) {
            tdm_load_w_tile(Wg + (size_t)dBase * C_DIM + k0, (void*)Wf);
            __builtin_amdgcn_s_wait_tensorcnt(0);
        }
#else
#pragma unroll
        for (int i = 0; i < 4; ++i) {
            int id  = i * 256 + tid;          // 0..1023 float4 slots
            int row = id >> 3;                // 0..127
            int c4  = (id & 7) << 2;          // 0,4,...,28
            float4 w4 = *(const float4*)(Wg + (size_t)(dBase + row) * C_DIM + k0 + c4);
            *(float4*)(&Wf[row * LDW + c4]) = w4;
        }
#endif
        // ---- stage X tile transposed: Xt[p][c], 32 c-rows x 128 p ----------
#pragma unroll
        for (int i = 0; i < 4; ++i) {
            int id = i * 256 + tid;
            int c  = id >> 5;                 // 0..31
            int p4 = (id & 31) << 2;          // 0,4,...,124
            float4 x4 = *(const float4*)(Ximg + (size_t)(k0 + c) * HW + pBase + p4);
            Xt[(p4 + 0) * LDT + c] = (__bf16)x4.x;
            Xt[(p4 + 1) * LDT + c] = (__bf16)x4.y;
            Xt[(p4 + 2) * LDT + c] = (__bf16)x4.z;
            Xt[(p4 + 3) * LDT + c] = (__bf16)x4.w;
        }
        __syncthreads();

        // ---- A fragment (16x32 bf16): lane<16: M=lane, K {0..7,16..23};
        //      lane>=16: M=lane-16, K {8..15,24..31}; convert f32->bf16 ------
        const float* wr = &Wf[(wv * 16 + l16) * LDW];
        const float4 f0 = *(const float4*)(wr + 8 * half);
        const float4 f1 = *(const float4*)(wr + 8 * half + 4);
        const float4 f2 = *(const float4*)(wr + 8 * half + 16);
        const float4 f3 = *(const float4*)(wr + 8 * half + 20);
        v16bf A;
        A[0]  = (__bf16)f0.x; A[1]  = (__bf16)f0.y; A[2]  = (__bf16)f0.z; A[3]  = (__bf16)f0.w;
        A[4]  = (__bf16)f1.x; A[5]  = (__bf16)f1.y; A[6]  = (__bf16)f1.z; A[7]  = (__bf16)f1.w;
        A[8]  = (__bf16)f2.x; A[9]  = (__bf16)f2.y; A[10] = (__bf16)f2.z; A[11] = (__bf16)f2.w;
        A[12] = (__bf16)f3.x; A[13] = (__bf16)f3.y; A[14] = (__bf16)f3.z; A[15] = (__bf16)f3.w;

        // ---- 8 p-subtiles: B fragment (32x16 bf16): lane<16: N=lane,K {0..7,16..23};
        //      lane>=16: N=lane-16, K {8..15,24..31} ---------------------------------
#pragma unroll
        for (int s = 0; s < 8; ++s) {
            const __bf16* xrow = &Xt[(s * 16 + l16) * LDT + 8 * half];
            v8bf b0 = *(const v8bf*)(xrow);
            v8bf b1 = *(const v8bf*)(xrow + 16);
            v16bf B = __builtin_shufflevector(b0, b1,
                          0, 1, 2, 3, 4, 5, 6, 7, 8, 9, 10, 11, 12, 13, 14, 15);
            acc[s] = __builtin_amdgcn_wmma_f32_16x16x32_bf16(
                         false, A, false, B, (short)0, acc[s], false, false);
        }
        __syncthreads();
    }

    // ---- epilogue: D layout VGPR r -> M = r + 8*half, N = lane&15 ----------
    const int dWave = dBase + wv * 16;
#pragma unroll
    for (int r = 0; r < 8; ++r) {
        const int   d  = dWave + r + 8 * half;
        const float bv = bias[d];
#pragma unroll
        for (int s = 0; s < 8; ++s) {
            const int   p   = pBase + s * 16 + l16;
            const float val = (acc[s][r] + bv) * scale;
            if (EPI == 0)
                Ybf[((size_t)img * C_DIM + d) * HW + p] = (__bf16)val;
            else
                Yf[((size_t)p * 2 + img) * C_DIM + d] = val;
        }
    }
}

// ---------------------------------------------------------------------------
// Per-(b,i,j,c) windowed cross-view attention over NK=16 keys, L=2304.
// scores[m] = <q_win, k_win[m]>; softmax over m; out = sum_m a[m]*v_win[m].
// Output stitched directly into local[b, c, h, w] (f32).
// ---------------------------------------------------------------------------
__global__ __launch_bounds__(256) void attn_softmax(
    const __bf16* __restrict__ Yq, const __bf16* __restrict__ Yk,
    const __bf16* __restrict__ Yv, float* __restrict__ local)
{
    __shared__ float red[16 * 8];
    __shared__ float sc[16];

    const int tid  = threadIdx.x;
    const int wv   = tid >> 5;
    const int lane = tid & 31;

    const int bidx = blockIdx.x;          // 2048 = b(2) * i(2) * j(2) * c(256)
    const int c = bidx & 255;
    const int j = (bidx >> 8) & 1;
    const int i = (bidx >> 9) & 1;
    const int b = bidx >> 10;

    const size_t qbase   = ((size_t)b * C_DIM + c) * HW;
    const int    winBase = i * 48 * 96 + j * 48;

    float s[16];
#pragma unroll
    for (int m = 0; m < 16; ++m) s[m] = 0.f;

    for (int l = tid; l < 2304; l += 256) {
        const int hh  = l / 48;
        const int ww  = l - hh * 48;
        const int pix = winBase + hh * 96 + ww;
        const float qv = (float)Yq[qbase + pix];
#pragma unroll
        for (int m = 0; m < 16; ++m) {
            const size_t kb = ((size_t)(b * 16 + m) * C_DIM + c) * HW;
            s[m] += qv * (float)Yk[kb + pix];
        }
    }
#pragma unroll
    for (int m = 0; m < 16; ++m) {
#pragma unroll
        for (int o = 16; o > 0; o >>= 1) s[m] += __shfl_down(s[m], o);
    }
    if (lane == 0) {
#pragma unroll
        for (int m = 0; m < 16; ++m) red[m * 8 + wv] = s[m];
    }
    __syncthreads();
    if (tid < 16) {
        float t = 0.f;
        for (int r = 0; r < 8; ++r) t += red[tid * 8 + r];
        sc[tid] = t;
    }
    __syncthreads();

    float mx = sc[0];
#pragma unroll
    for (int m = 1; m < 16; ++m) mx = fmaxf(mx, sc[m]);
    float a[16];
    float sum = 0.f;
#pragma unroll
    for (int m = 0; m < 16; ++m) { a[m] = __expf(sc[m] - mx); sum += a[m]; }
    const float inv = 1.f / sum;
#pragma unroll
    for (int m = 0; m < 16; ++m) a[m] *= inv;

    for (int l = tid; l < 2304; l += 256) {
        const int hh  = l / 48;
        const int ww  = l - hh * 48;
        const int pix = winBase + hh * 96 + ww;
        float o = 0.f;
#pragma unroll
        for (int m = 0; m < 16; ++m) {
            const size_t vb = ((size_t)(b * 16 + m) * C_DIM + c) * HW;
            o += a[m] * (float)Yv[vb + pix];
        }
        local[((size_t)b * C_DIM + c) * HW + pix] = o;
    }
}

// ---------------------------------------------------------------------------
extern "C" void kernel_launch(void* const* d_in, const int* in_sizes, int n_in,
                              void* d_out, int out_size, void* d_ws, size_t ws_size,
                              hipStream_t stream) {
    const float* q  = (const float*)d_in[0];
    const float* k  = (const float*)d_in[1];
    const float* v  = (const float*)d_in[2];
    const float* Wq = (const float*)d_in[3];
    const float* bq = (const float*)d_in[4];
    const float* Wk = (const float*)d_in[5];
    const float* bk = (const float*)d_in[6];
    const float* Wv = (const float*)d_in[7];
    const float* bv = (const float*)d_in[8];
    const float* Wp = (const float*)d_in[9];
    const float* bp = (const float*)d_in[10];
    float* out = (float*)d_out;

    char* ws = (char*)d_ws;
    const size_t SZ_Q  = (size_t)2  * C_DIM * HW * sizeof(unsigned short); //  9.4 MB
    const size_t SZ_KV = (size_t)32 * C_DIM * HW * sizeof(unsigned short); //  151 MB
    __bf16* Yq    = (__bf16*)(ws);
    __bf16* Yk    = (__bf16*)(ws + SZ_Q);
    __bf16* Yv    = (__bf16*)(ws + SZ_Q + SZ_KV);
    float*  local = (float*) (ws + SZ_Q + 2 * SZ_KV);

    const dim3 blk(256);
    // q projection (+bias, /sqrt(256)=1/16), 2 images
    gemm_wmma<0><<<dim3(72, 2, 2),  blk, 0, stream>>>(q, Wq, bq, Yq, nullptr, 0.0625f);
    // k / v projections, 32 images each
    gemm_wmma<0><<<dim3(72, 2, 32), blk, 0, stream>>>(k, Wk, bk, Yk, nullptr, 1.0f);
    gemm_wmma<0><<<dim3(72, 2, 32), blk, 0, stream>>>(v, Wv, bv, Yv, nullptr, 1.0f);
    // windowed per-channel attention -> stitched local [b, c, h, w]
    attn_softmax<<<dim3(2048), blk, 0, stream>>>(Yq, Yk, Yv, local);
    // output projection, store as [hw, bs, c]
    gemm_wmma<1><<<dim3(72, 2, 2),  blk, 0, stream>>>(local, Wp, bp, nullptr, out, 1.0f);
}
